// MambaLayerOnlyspiral_51221779972677
// MI455X (gfx1250) — matrix operand order repaired and checked by reading, not compiled, not run on hardware
//
#include <hip/hip_runtime.h>

// ---------------- sizes ----------------
#define BSZ    2
#define LTOK   4096
#define CDIM   256
#define DIN    512
#define NPROJ  1024   // 2*DIN
#define DSTATE 16
#define DTRANK 16
#define XDBL   48     // DTRANK + 2*DSTATE
#define NCHUNK 16
#define CHUNK  256    // LTOK / NCHUNK

typedef __bf16 bf16_t;
typedef __attribute__((ext_vector_type(16))) __bf16 v16bf;
typedef __attribute__((ext_vector_type(8)))  float  v8f;
typedef __attribute__((ext_vector_type(4)))  int    v4i;

// ---------------- gfx1250 async-to-LDS helpers (guarded) ----------------
#if defined(__AMDGCN__) && defined(__gfx1250__)
#  if __has_builtin(__builtin_amdgcn_global_load_async_to_lds_b128)
#    define HAVE_ASYNC_LDS 1
#  endif
#endif

__device__ __forceinline__ void async_cp16(const void* g, void* l) {
#ifdef HAVE_ASYNC_LDS
  __builtin_amdgcn_global_load_async_to_lds_b128(
      (__attribute__((address_space(1))) v4i*)g,
      (__attribute__((address_space(3))) v4i*)l, 0, 0);
#else
  *(uint4*)l = *(const uint4*)g;
#endif
}

__device__ __forceinline__ void async_wait_all() {
#ifdef HAVE_ASYNC_LDS
#  if __has_builtin(__builtin_amdgcn_s_wait_asynccnt)
  __builtin_amdgcn_s_wait_asynccnt(0);
#  else
  asm volatile("s_wait_asynccnt 0x0" ::: "memory");
#  endif
#endif
}

// =====================================================================
// 0. permutation index extraction: one wave per row, argmax of one-hot
// =====================================================================
__global__ void perm_kernel(const float* __restrict__ e0,
                            const float* __restrict__ e1,
                            const float* __restrict__ e2,
                            int* __restrict__ p0, int* __restrict__ p1,
                            int* __restrict__ p2) {
  int tid  = blockIdx.x * blockDim.x + threadIdx.x;
  int wid  = tid >> 5;
  int lane = tid & 31;
  if (wid >= 3 * LTOK) return;
  int mat = wid / LTOK, r = wid % LTOK;
  const float* row = (mat == 0 ? e0 : mat == 1 ? e1 : e2) + (size_t)r * LTOK;
  int idx = -1;
  for (int j = lane; j < LTOK; j += 32)
    if (row[j] > 0.5f) idx = j;
  for (int m = 16; m >= 1; m >>= 1) {
    int o = __shfl_xor(idx, m, 32);
    idx = idx > o ? idx : o;
  }
  if (lane == 0) (mat == 0 ? p0 : mat == 1 ? p1 : p2)[r] = idx;
}

// =====================================================================
// 1. gather (spiral) + LayerNorm over channels -> bf16 tokens [B,L,C]
// =====================================================================
__global__ void ln_gather_kernel(const float* __restrict__ x,
                                 const int* __restrict__ ps,
                                 const float* __restrict__ gamma,
                                 const float* __restrict__ beta,
                                 bf16_t* __restrict__ uln) {
  int gid = blockIdx.x;          // b*L + i
  int b = gid >> 12, i = gid & (LTOK - 1);
  int c = threadIdx.x;           // 256 threads
  int j = ps[i];
  float v = x[((size_t)(b * CDIM + c)) * LTOK + j];
  __shared__ float red[CDIM];
  red[c] = v;
  __syncthreads();
  for (int off = 128; off; off >>= 1) {
    if (c < off) red[c] += red[c + off];
    __syncthreads();
  }
  float mu = red[0] * (1.f / CDIM);
  __syncthreads();
  float dv = v - mu;
  red[c] = dv * dv;
  __syncthreads();
  for (int off = 128; off; off >>= 1) {
    if (c < off) red[c] += red[c + off];
    __syncthreads();
  }
  float var = red[0] * (1.f / CDIM);
  float o = dv * rsqrtf(var + 1e-5f) * gamma[c] + beta[c];
  uln[((size_t)(b * LTOK + i)) * CDIM + c] = (bf16_t)o;
}

// =====================================================================
// 2. weight prep
// =====================================================================
__global__ void cvt_wt_kernel(const float* __restrict__ ip, bf16_t* __restrict__ wt) {
  int idx = blockIdx.x * blockDim.x + threadIdx.x;
  if (idx >= NPROJ * CDIM) return;
  int n = idx / CDIM, k = idx % CDIM;
  wt[(size_t)n * CDIM + k] = (bf16_t)ip[(size_t)k * NPROJ + n];
}

// fused W[o][d] = sum_c op[d][c] * cw[o][coff + c]  (bf16, [256][512])
__global__ void fuse_kernel(const float* __restrict__ op, const float* __restrict__ cw,
                            int coff, bf16_t* __restrict__ wft) {
  int idx = blockIdx.x * blockDim.x + threadIdx.x;
  if (idx >= CDIM * DIN) return;
  int o = idx / DIN, d = idx % DIN;
  float acc = 0.f;
  const float* opr = op + (size_t)d * CDIM;
  const float* cwr = cw + (size_t)o * DIN + coff;
  for (int c = 0; c < CDIM; ++c) acc += opr[c] * cwr[c];
  wft[(size_t)o * DIN + d] = (bf16_t)acc;
}

__global__ void aneg_kernel(const float* __restrict__ alog, float* __restrict__ an) {
  int idx = blockIdx.x * blockDim.x + threadIdx.x;
  if (idx >= DIN * DSTATE) return;
  an[idx] = -__expf(alog[idx]);
}

// =====================================================================
// 3. in_proj GEMM: [8192 x 256] bf16 @ [256 x 1024] -> xz f32
//    A tile staged once per block into LDS via async-to-LDS; WMMA bf16.
// =====================================================================
__global__ void in_proj_wmma_kernel(const bf16_t* __restrict__ uln,
                                    const bf16_t* __restrict__ wt,  // [1024][256]
                                    float* __restrict__ xz, int rev) {
  __shared__ __align__(16) bf16_t atile[16 * CDIM];   // 8 KB
  const int tid  = threadIdx.x;
  const int lane = tid & 31;
  const int wave = tid >> 5;
  const int m16 = lane & 15, g = lane >> 4;
  const int mbase = blockIdx.x * 16;
  const int nbase = blockIdx.y * 64 + wave * 16;
  const int b0 = mbase >> 12;                 // tile never spans batch

  // stage A tile: 16 rows x 512B = 512 x 16B chunks, 128 threads x 4
#pragma unroll
  for (int j = 0; j < 4; ++j) {
    int c = tid + 128 * j;                    // 0..511
    int r = c >> 5;                           // row 0..15
    int off = (c & 31) * 8;                   // bf16 elements (16B chunks)
    int tok = (mbase + r) & (LTOK - 1);
    int stok = rev ? (LTOK - 1 - tok) : tok;
    async_cp16(uln + ((size_t)(b0 * LTOK + stok)) * CDIM + off,
               &atile[r * CDIM + off]);
  }
  async_wait_all();
  __syncthreads();

  const bf16_t* arow = &atile[m16 * CDIM];
  const bf16_t* brow = wt + (size_t)(nbase + m16) * CDIM;

  v8f acc = {0.f, 0.f, 0.f, 0.f, 0.f, 0.f, 0.f, 0.f};
  for (int kb = 0; kb < CDIM; kb += 32) {
    v16bf a, bm;
    uint4* au = reinterpret_cast<uint4*>(&a);
    au[0] = *(const uint4*)(arow + kb + g * 8);        // ds_load_b128
    au[1] = *(const uint4*)(arow + kb + 16 + g * 8);
    uint4* bu = reinterpret_cast<uint4*>(&bm);
    bu[0] = *(const uint4*)(brow + kb + g * 8);
    bu[1] = *(const uint4*)(brow + kb + 16 + g * 8);
    acc = __builtin_amdgcn_wmma_f32_16x16x32_bf16(false, a, false, bm,
                                                  (short)0, acc, false, false);
  }
#pragma unroll
  for (int r = 0; r < 8; ++r) {
    int orow = mbase + r + 8 * g;
    int ob = orow >> 12, otok = orow & (LTOK - 1);
    xz[((size_t)(ob * LTOK + otok)) * NPROJ + nbase + m16] = acc[r];
  }
}

// =====================================================================
// 4. causal depthwise conv (k=4) + SiLU
// =====================================================================
__global__ void dwconv_kernel(const float* __restrict__ xz,
                              const float* __restrict__ cw,
                              const float* __restrict__ cb,
                              float* __restrict__ xc) {
  size_t i = (size_t)blockIdx.x * blockDim.x + threadIdx.x;
  if (i >= (size_t)BSZ * LTOK * DIN) return;
  size_t row = i / DIN;
  int d = (int)(i % DIN);
  int b = (int)(row / LTOK), l = (int)(row % LTOK);
  float acc = cb[d];
#pragma unroll
  for (int k = 0; k < 4; ++k) {
    int ll = l + k - 3;
    if (ll >= 0) acc += xz[((size_t)(b * LTOK + ll)) * NPROJ + d] * cw[d * 4 + k];
  }
  xc[i] = acc / (1.f + __expf(-acc));   // silu
}

// =====================================================================
// 5. x_proj: [rows x 512] @ [512 x 48]
// =====================================================================
__global__ void xproj_kernel(const float* __restrict__ xc,
                             const float* __restrict__ xp,
                             float* __restrict__ xdbl) {
  int idx = blockIdx.x * blockDim.x + threadIdx.x;
  if (idx >= BSZ * LTOK * XDBL) return;
  int row = idx / XDBL, j = idx % XDBL;
  const float* xr = xc + (size_t)row * DIN;
  float acc = 0.f;
  for (int d = 0; d < DIN; ++d) acc += xr[d] * xp[d * XDBL + j];
  xdbl[idx] = acc;
}

// =====================================================================
// 6. delta = softplus(dt @ dt_w + dt_b)
// =====================================================================
__global__ void delta_kernel(const float* __restrict__ xdbl,
                             const float* __restrict__ dtw,
                             const float* __restrict__ dtb,
                             float* __restrict__ delta) {
  size_t idx = (size_t)blockIdx.x * blockDim.x + threadIdx.x;
  if (idx >= (size_t)BSZ * LTOK * DIN) return;
  size_t row = idx / DIN;
  int d = (int)(idx % DIN);
  const float* xr = xdbl + row * XDBL;
  float acc = dtb[d];
#pragma unroll
  for (int r = 0; r < DTRANK; ++r) acc += xr[r] * dtw[r * DIN + d];
  delta[idx] = (acc > 20.f) ? acc : log1pf(__expf(acc));
}

// =====================================================================
// 7. chunked selective scan (exact): 16 chunks of 256 steps
//    pass1: per-chunk decay product P and partial end state H (h0 = 0)
//    combine: 16-step sequential prefix -> exact chunk initial states H0
//    pass2: replay each chunk from exact H0, emit y
//    Task = (b, d, chunk) per 16-lane half-wave, state s per lane.
// =====================================================================
__global__ void scan_part1_kernel(const float* __restrict__ delta,
                                  const float* __restrict__ xc,
                                  const float* __restrict__ xdbl,
                                  const float* __restrict__ aneg,
                                  float* __restrict__ Pc, float* __restrict__ Hc) {
  int tid = blockIdx.x * blockDim.x + threadIdx.x;
  int hw = tid >> 4;
  int s = tid & 15;
  if (hw >= BSZ * NCHUNK * DIN) return;
  int d = hw % DIN;
  int c = (hw / DIN) % NCHUNK;
  int b = hw / (DIN * NCHUNK);
  float Av = aneg[d * DSTATE + s];
  float h = 0.f, p = 1.f;
  size_t base_d = (size_t)b * LTOK * DIN + d;
  size_t base_x = (size_t)b * LTOK * XDBL;
  int l0 = c * CHUNK;
  for (int i = 0; i < CHUNK; ++i) {
    int l = l0 + i;
    float dt = delta[base_d + (size_t)l * DIN];
    float xv = xc[base_d + (size_t)l * DIN];
    float Bv = xdbl[base_x + (size_t)l * XDBL + DTRANK + s];
    float a = __expf(dt * Av);
    h = h * a + dt * xv * Bv;
    p *= a;
    if (i + 8 < CHUNK)
      __builtin_prefetch(&xdbl[base_x + (size_t)(l + 8) * XDBL], 0, 0);
  }
  size_t o = (size_t)hw * DSTATE + s;
  Pc[o] = p;
  Hc[o] = h;
}

__global__ void scan_combine_kernel(const float* __restrict__ Pc,
                                    const float* __restrict__ Hc,
                                    float* __restrict__ H0) {
  int tid = blockIdx.x * blockDim.x + threadIdx.x;
  if (tid >= BSZ * DIN * DSTATE) return;
  int s = tid % DSTATE;
  int d = (tid / DSTATE) % DIN;
  int b = tid / (DSTATE * DIN);
  float h0 = 0.f;
  for (int c = 0; c < NCHUNK; ++c) {
    size_t o = ((((size_t)b * NCHUNK + c) * DIN) + d) * DSTATE + s;
    H0[o] = h0;
    h0 = Pc[o] * h0 + Hc[o];
  }
}

__global__ void scan_part2_kernel(const float* __restrict__ delta,
                                  const float* __restrict__ xc,
                                  const float* __restrict__ xdbl,
                                  const float* __restrict__ aneg,
                                  const float* __restrict__ dvec,
                                  const float* __restrict__ H0,
                                  float* __restrict__ y) {
  int tid = blockIdx.x * blockDim.x + threadIdx.x;
  int hw = tid >> 4;
  int s = tid & 15;
  if (hw >= BSZ * NCHUNK * DIN) return;
  int d = hw % DIN;
  int c = (hw / DIN) % NCHUNK;
  int b = hw / (DIN * NCHUNK);
  float Av = aneg[d * DSTATE + s];
  float Dd = dvec[d];
  float h = H0[(size_t)hw * DSTATE + s];
  size_t base_d = (size_t)b * LTOK * DIN + d;
  size_t base_x = (size_t)b * LTOK * XDBL;
  int l0 = c * CHUNK;
  for (int i = 0; i < CHUNK; ++i) {
    int l = l0 + i;
    float dt = delta[base_d + (size_t)l * DIN];
    float xv = xc[base_d + (size_t)l * DIN];
    float Bv = xdbl[base_x + (size_t)l * XDBL + DTRANK + s];
    float Cv = xdbl[base_x + (size_t)l * XDBL + DTRANK + DSTATE + s];
    float a = __expf(dt * Av);
    h = h * a + dt * xv * Bv;
    float cc = h * Cv;
    cc += __shfl_xor(cc, 8, 32);
    cc += __shfl_xor(cc, 4, 32);
    cc += __shfl_xor(cc, 2, 32);
    cc += __shfl_xor(cc, 1, 32);
    if (s == 0) y[base_d + (size_t)l * DIN] = cc + xv * Dd;
    if (i + 8 < CHUNK)
      __builtin_prefetch(&xdbl[base_x + (size_t)(l + 8) * XDBL], 0, 0);
  }
}

// =====================================================================
// 8. gating: ym = y * silu(z)  -> bf16
// =====================================================================
__global__ void gate_kernel(const float* __restrict__ y,
                            const float* __restrict__ xz,
                            bf16_t* __restrict__ ym) {
  size_t i = (size_t)blockIdx.x * blockDim.x + threadIdx.x;
  if (i >= (size_t)BSZ * LTOK * DIN) return;
  size_t row = i / DIN;
  int d = (int)(i % DIN);
  float z = xz[row * NPROJ + DIN + d];
  float g = z / (1.f + __expf(-z));
  ym[i] = (bf16_t)(y[i] * g);
}

// =====================================================================
// 9. final: out[b,o,i] = ym1[b,pd[i],:]·W1[:,o] + ym2[b,pdr[i],:]·W2[:,o] + cb[o]
//    Gathered A rows staged to LDS via async-to-LDS, WMMA bf16.
// =====================================================================
__global__ void final_wmma_kernel(const bf16_t* __restrict__ ym1,
                                  const bf16_t* __restrict__ ym2,
                                  const bf16_t* __restrict__ w1t,  // [256][512]
                                  const bf16_t* __restrict__ w2t,  // [256][512]
                                  const int* __restrict__ pd,
                                  const int* __restrict__ pdr,
                                  const float* __restrict__ convb,
                                  float* __restrict__ out) {
  __shared__ __align__(16) bf16_t s1[16 * DIN];   // 16 KB
  __shared__ __align__(16) bf16_t s2[16 * DIN];   // 16 KB
  const int tid  = threadIdx.x;
  const int lane = tid & 31;
  const int wave = tid >> 5;
  const int m16 = lane & 15, g = lane >> 4;
  const int b = blockIdx.z;
  const int mbase = blockIdx.x * 16;              // spatial i tile
  const int nbase = blockIdx.y * 64 + wave * 16;  // out-channel tile

  // stage 2 gathered tiles: 16 rows x 1KB each = 1024 x 16B chunks per tile
#pragma unroll
  for (int j = 0; j < 8; ++j) {
    int c = tid + 128 * j;                         // 0..1023
    int r = c >> 6;                                // row 0..15
    int off = (c & 63) * 8;                        // bf16 elements
    int t1r = pd[mbase + r];
    int t2r = pdr[mbase + r];
    async_cp16(ym1 + ((size_t)(b * LTOK + t1r)) * DIN + off, &s1[r * DIN + off]);
    async_cp16(ym2 + ((size_t)(b * LTOK + t2r)) * DIN + off, &s2[r * DIN + off]);
  }
  async_wait_all();
  __syncthreads();

  const bf16_t* a1 = &s1[m16 * DIN];
  const bf16_t* a2 = &s2[m16 * DIN];
  const bf16_t* b1 = w1t + (size_t)(nbase + m16) * DIN;
  const bf16_t* b2 = w2t + (size_t)(nbase + m16) * DIN;

  v8f acc = {0.f, 0.f, 0.f, 0.f, 0.f, 0.f, 0.f, 0.f};
  for (int kb = 0; kb < DIN; kb += 32) {
    v16bf a, bm;
    uint4* au = reinterpret_cast<uint4*>(&a);
    au[0] = *(const uint4*)(a1 + kb + g * 8);
    au[1] = *(const uint4*)(a1 + kb + 16 + g * 8);
    uint4* bu = reinterpret_cast<uint4*>(&bm);
    bu[0] = *(const uint4*)(b1 + kb + g * 8);
    bu[1] = *(const uint4*)(b1 + kb + 16 + g * 8);
    acc = __builtin_amdgcn_wmma_f32_16x16x32_bf16(false, a, false, bm,
                                                  (short)0, acc, false, false);
  }
  for (int kb = 0; kb < DIN; kb += 32) {
    v16bf a, bm;
    uint4* au = reinterpret_cast<uint4*>(&a);
    au[0] = *(const uint4*)(a2 + kb + g * 8);
    au[1] = *(const uint4*)(a2 + kb + 16 + g * 8);
    uint4* bu = reinterpret_cast<uint4*>(&bm);
    bu[0] = *(const uint4*)(b2 + kb + g * 8);
    bu[1] = *(const uint4*)(b2 + kb + 16 + g * 8);
    acc = __builtin_amdgcn_wmma_f32_16x16x32_bf16(false, a, false, bm,
                                                  (short)0, acc, false, false);
  }
  int o = nbase + m16;
  float bias = convb[o];
#pragma unroll
  for (int r = 0; r < 8; ++r) {
    int ii = mbase + r + 8 * g;
    out[((size_t)(b * CDIM + o)) * LTOK + ii] = acc[r] + bias;
  }
}

// =====================================================================
// launch
// =====================================================================
extern "C" void kernel_launch(void* const* d_in, const int* in_sizes, int n_in,
                              void* d_out, int out_size, void* d_ws, size_t ws_size,
                              hipStream_t stream) {
  (void)in_sizes; (void)n_in; (void)out_size; (void)ws_size;
  const float* x        = (const float*)d_in[0];
  const float* spiral   = (const float*)d_in[1];
  const float* despir   = (const float*)d_in[2];
  const float* despir_r = (const float*)d_in[3];
  const float* gamma    = (const float*)d_in[4];
  const float* beta     = (const float*)d_in[5];
  const float* m_in_proj[2]  = {(const float*)d_in[6],  (const float*)d_in[15]};
  const float* m_conv_w[2]   = {(const float*)d_in[7],  (const float*)d_in[16]};
  const float* m_conv_b[2]   = {(const float*)d_in[8],  (const float*)d_in[17]};
  const float* m_x_proj[2]   = {(const float*)d_in[9],  (const float*)d_in[18]};
  const float* m_dt_w[2]     = {(const float*)d_in[10], (const float*)d_in[19]};
  const float* m_dt_b[2]     = {(const float*)d_in[11], (const float*)d_in[20]};
  const float* m_A_log[2]    = {(const float*)d_in[12], (const float*)d_in[21]};
  const float* m_D[2]        = {(const float*)d_in[13], (const float*)d_in[22]};
  const float* m_out_proj[2] = {(const float*)d_in[14], (const float*)d_in[23]};
  const float* cw_final = (const float*)d_in[24];
  const float* cb_final = (const float*)d_in[25];
  float* out = (float*)d_out;

  // workspace carve-up
  char* w = (char*)d_ws;
  auto alloc = [&](size_t bytes) -> void* {
    void* p = (void*)w;
    w += (bytes + 255) & ~(size_t)255;
    return p;
  };
  int* perm_s  = (int*)alloc(LTOK * 4);
  int* perm_d  = (int*)alloc(LTOK * 4);
  int* perm_dr = (int*)alloc(LTOK * 4);
  bf16_t* uln  = (bf16_t*)alloc((size_t)BSZ * LTOK * CDIM * 2);
  bf16_t* wt_in[2]  = {(bf16_t*)alloc((size_t)NPROJ * CDIM * 2),
                       (bf16_t*)alloc((size_t)NPROJ * CDIM * 2)};
  bf16_t* wf_t[2]   = {(bf16_t*)alloc((size_t)CDIM * DIN * 2),
                       (bf16_t*)alloc((size_t)CDIM * DIN * 2)};
  float*  aneg[2]   = {(float*)alloc((size_t)DIN * DSTATE * 4),
                       (float*)alloc((size_t)DIN * DSTATE * 4)};
  bf16_t* ym[2]     = {(bf16_t*)alloc((size_t)BSZ * LTOK * DIN * 2),
                       (bf16_t*)alloc((size_t)BSZ * LTOK * DIN * 2)};
  // big shared scratch (reused by m1 then m2)
  float* xz    = (float*)alloc((size_t)BSZ * LTOK * NPROJ * 4);
  float* xc    = (float*)alloc((size_t)BSZ * LTOK * DIN * 4);
  float* xdbl  = (float*)alloc((size_t)BSZ * LTOK * XDBL * 4);
  float* delta = (float*)alloc((size_t)BSZ * LTOK * DIN * 4);
  float* ybuf  = (float*)alloc((size_t)BSZ * LTOK * DIN * 4);
  // chunked-scan state
  float* Pc = (float*)alloc((size_t)BSZ * NCHUNK * DIN * DSTATE * 4);
  float* Hc = (float*)alloc((size_t)BSZ * NCHUNK * DIN * DSTATE * 4);
  float* H0 = (float*)alloc((size_t)BSZ * NCHUNK * DIN * DSTATE * 4);

  // ---- prep ----
  perm_kernel<<<(3 * LTOK * 32 + 127) / 128, 128, 0, stream>>>(
      spiral, despir, despir_r, perm_s, perm_d, perm_dr);
  ln_gather_kernel<<<BSZ * LTOK, CDIM, 0, stream>>>(x, perm_s, gamma, beta, uln);
  for (int m = 0; m < 2; ++m) {
    cvt_wt_kernel<<<(NPROJ * CDIM + 255) / 256, 256, 0, stream>>>(m_in_proj[m], wt_in[m]);
    fuse_kernel<<<(CDIM * DIN + 255) / 256, 256, 0, stream>>>(
        m_out_proj[m], cw_final, m * CDIM, wf_t[m]);
    aneg_kernel<<<(DIN * DSTATE + 255) / 256, 256, 0, stream>>>(m_A_log[m], aneg[m]);
  }

  // ---- mamba passes ----
  const size_t nrowd = (size_t)BSZ * LTOK * DIN;
  const int scan_tasks = BSZ * NCHUNK * DIN * 16;   // threads for part1/part2
  for (int m = 0; m < 2; ++m) {
    dim3 gin(BSZ * LTOK / 16, NPROJ / 64, 1);
    in_proj_wmma_kernel<<<gin, 128, 0, stream>>>(uln, wt_in[m], xz, m);
    dwconv_kernel<<<(int)((nrowd + 255) / 256), 256, 0, stream>>>(
        xz, m_conv_w[m], m_conv_b[m], xc);
    xproj_kernel<<<(BSZ * LTOK * XDBL + 255) / 256, 256, 0, stream>>>(
        xc, m_x_proj[m], xdbl);
    delta_kernel<<<(int)((nrowd + 255) / 256), 256, 0, stream>>>(
        xdbl, m_dt_w[m], m_dt_b[m], delta);
    scan_part1_kernel<<<(scan_tasks + 255) / 256, 256, 0, stream>>>(
        delta, xc, xdbl, aneg[m], Pc, Hc);
    scan_combine_kernel<<<(BSZ * DIN * DSTATE + 255) / 256, 256, 0, stream>>>(
        Pc, Hc, H0);
    scan_part2_kernel<<<(scan_tasks + 255) / 256, 256, 0, stream>>>(
        delta, xc, xdbl, aneg[m], m_D[m], H0, ybuf);
    gate_kernel<<<(int)((nrowd + 255) / 256), 256, 0, stream>>>(ybuf, xz, ym[m]);
  }

  // ---- fused de-permute + concat + 1x1 conv ----
  dim3 gfin(LTOK / 16, CDIM / 64, BSZ);
  final_wmma_kernel<<<gfin, 128, 0, stream>>>(
      ym[0], ym[1], wf_t[0], wf_t[1], perm_d, perm_dr, cb_final, out);
}
